// DenseCausalSelfAttention_35304631173151
// MI455X (gfx1250) — matrix operand order repaired and checked by reading, not compiled, not run on hardware
//
#include <hip/hip_runtime.h>

typedef __attribute__((ext_vector_type(16))) __bf16 v16bf;
typedef __attribute__((ext_vector_type(8)))  float  v8f;
typedef __attribute__((ext_vector_type(4)))  unsigned int u32x4;
typedef int v4i_g __attribute__((vector_size(16)));  // GCC-vector int4 for builtin signature

union AF {
  u32x4 u[2];
  v16bf v;
  unsigned short s[16];
};

__device__ __forceinline__ unsigned short f32_to_bf16_rne(float f) {
  union { float f; unsigned int u; } x;
  x.f = f;
  unsigned int u = x.u;
  unsigned int r = (u + 0x7FFFu + ((u >> 16) & 1u)) >> 16;
  if ((u & 0x7F800000u) == 0x7F800000u) r = u >> 16;  // inf/nan: truncate
  return (unsigned short)r;
}

// ---------------------------------------------------------------------------
// gfx1250 async global->LDS copy (16B per lane), ASYNCcnt-tracked.
// ---------------------------------------------------------------------------
__device__ __forceinline__ void async_g2l_b128(const void* g, void* l) {
#if __has_builtin(__builtin_amdgcn_global_load_async_to_lds_b128)
  __builtin_amdgcn_global_load_async_to_lds_b128(
      (__attribute__((address_space(1))) v4i_g*)(g),
      (__attribute__((address_space(3))) v4i_g*)(l), 0, 0);
#else
  unsigned int laddr =
      (unsigned int)(unsigned long long)(__attribute__((address_space(3))) void*)l;
  unsigned long long gaddr = (unsigned long long)g;
  asm volatile("global_load_async_to_lds_b128 %0, %1, off"
               :: "v"(laddr), "v"(gaddr)
               : "memory");
#endif
}

__device__ __forceinline__ void wait_async0() {
#if __has_builtin(__builtin_amdgcn_s_wait_asynccnt)
  __builtin_amdgcn_s_wait_asynccnt(0);
#else
  asm volatile("s_wait_asynccnt 0x0" ::: "memory");
#endif
}

// ---------------------------------------------------------------------------
// Generic bf16 GEMM: C(f32, MxN) = A(bf16, MxK) * B(bf16, KxN), row-major.
// Block tile 128x64, k-step 32, 256 threads (8 waves), each wave 32x32 out.
// A tile staged with async global->LDS; B tile transposed through VGPRs.
// ---------------------------------------------------------------------------
__global__ __launch_bounds__(256) void gemm_bf16_kernel(
    const unsigned short* __restrict__ A,
    const unsigned short* __restrict__ Bw,
    float* __restrict__ C,
    int M, int N, int K) {
  __shared__ __attribute__((aligned(16))) unsigned short As[128 * 40];
  __shared__ __attribute__((aligned(16))) unsigned short Bt[64 * 40];

  const int tid  = threadIdx.x;
  const int lane = tid & 31;
  const int wave = tid >> 5;
  const int hl   = lane >> 4;   // half-of-wave (0/1)
  const int ll   = lane & 15;
  const int n0 = blockIdx.x * 64;
  const int m0 = blockIdx.y * 128;
  const int wm = wave >> 1;     // 0..3
  const int wn = wave & 1;      // 0..1

  v8f zero8;
#pragma unroll
  for (int e = 0; e < 8; ++e) zero8[e] = 0.0f;
  v8f acc[2][2];
#pragma unroll
  for (int i = 0; i < 2; ++i)
#pragma unroll
    for (int j = 0; j < 2; ++j) acc[i][j] = zero8;

  const int ar = tid >> 1, ap = tid & 1;   // A stage: row, 16-half part
  const int bk = tid >> 3, bs = tid & 7;   // B stage: k-row, 8-col segment
  const unsigned short* arow = A + (size_t)(m0 + ar) * K + ap * 16;
  const unsigned short* brow = Bw + n0 + bs * 8;

  for (int k0 = 0; k0 < K; k0 += 32) {
    // async-stage A tile (no transpose needed)
    async_g2l_b128(arow + k0,     &As[ar * 40 + ap * 16]);
    async_g2l_b128(arow + k0 + 8, &As[ar * 40 + ap * 16 + 8]);
    // B tile: transpose through registers
    u32x4 bv = *(const u32x4*)(brow + (size_t)(k0 + bk) * N);
    const unsigned short* bp = (const unsigned short*)&bv;
#pragma unroll
    for (int j = 0; j < 8; ++j) Bt[(bs * 8 + j) * 40 + bk] = bp[j];
    if (k0 + 32 < K) {
      __builtin_prefetch(arow + k0 + 32, 0, 1);
      __builtin_prefetch(brow + (size_t)(k0 + 32 + bk) * N, 0, 1);
    }
    wait_async0();
    __syncthreads();

    AF a[2], b[2];
#pragma unroll
    for (int tm = 0; tm < 2; ++tm) {
      int row = wm * 32 + tm * 16 + ll;
      a[tm].u[0] = *(const u32x4*)&As[row * 40 + 8 * hl];
      a[tm].u[1] = *(const u32x4*)&As[row * 40 + 16 + 8 * hl];
    }
#pragma unroll
    for (int tn = 0; tn < 2; ++tn) {
      int col = wn * 32 + tn * 16 + ll;
      b[tn].u[0] = *(const u32x4*)&Bt[col * 40 + 8 * hl];
      b[tn].u[1] = *(const u32x4*)&Bt[col * 40 + 16 + 8 * hl];
    }
#pragma unroll
    for (int tm = 0; tm < 2; ++tm)
#pragma unroll
      for (int tn = 0; tn < 2; ++tn)
        acc[tm][tn] = __builtin_amdgcn_wmma_f32_16x16x32_bf16(
            false, a[tm].v, false, b[tn].v, (short)0, acc[tm][tn], false, false);
    __syncthreads();
  }

#pragma unroll
  for (int tm = 0; tm < 2; ++tm)
#pragma unroll
    for (int tn = 0; tn < 2; ++tn)
#pragma unroll
      for (int r = 0; r < 8; ++r) {
        int row = m0 + wm * 32 + tm * 16 + r + 8 * hl;
        int col = n0 + wn * 32 + tn * 16 + ll;
        C[(size_t)row * N + col] = acc[tm][tn][r];
      }
}

// ---------------------------------------------------------------------------
// f32 -> bf16 elementwise
// ---------------------------------------------------------------------------
__global__ void cvt_f32_bf16_kernel(const float* __restrict__ in,
                                    unsigned short* __restrict__ out,
                                    long long n) {
  long long i = (long long)blockIdx.x * blockDim.x + threadIdx.x;
  long long stride = (long long)gridDim.x * blockDim.x;
  for (; i < n; i += stride) out[i] = f32_to_bf16_rne(in[i]);
}

// ---------------------------------------------------------------------------
// RoPE (adjacent-pair) + permute (B,T,H,64) f32 -> (B,H,T,64) bf16
// ---------------------------------------------------------------------------
__global__ void rope_permute_kernel(const float* __restrict__ in,
                                    unsigned short* __restrict__ out,
                                    int H, int T, int do_rope, long long npairs) {
  long long idx = (long long)blockIdx.x * blockDim.x + threadIdx.x;
  long long stride = (long long)gridDim.x * blockDim.x;
  for (; idx < npairs; idx += stride) {
    int i = (int)(idx & 31);
    long long rest = idx >> 5;
    int h = (int)(rest % H); rest /= H;
    int t = (int)(rest % T);
    int b = (int)(rest / T);
    const float* src = in + (((size_t)b * T + t) * H + h) * 64 + 2 * i;
    float e = src[0], o = src[1];
    if (do_rope) {
      float inv = powf(500000.0f, -(float)(2 * i) * (1.0f / 64.0f));
      float ang = (float)t * inv;
      float s, c;
      sincosf(ang, &s, &c);
      float e2 = e * c - o * s;
      o = o * c + e * s;
      e = e2;
    }
    unsigned short* dst = out + (((size_t)b * H + h) * T + t) * 64 + 2 * i;
    dst[0] = f32_to_bf16_rne(e);
    dst[1] = f32_to_bf16_rne(o);
  }
}

// ---------------------------------------------------------------------------
// Flash-attention (causal, GQA 32q/8kv, D=64).
// Block: (b,h, 128 q-rows), 8 waves x 16-row q tiles, k-step 32.
// K tile staged with async global->LDS; V transposed through VGPRs.
// ---------------------------------------------------------------------------
__global__ __launch_bounds__(256) void attn_kernel(
    const unsigned short* __restrict__ Q,   // (B,32,T,64) bf16 (roped)
    const unsigned short* __restrict__ Kc,  // (B, 8,T,64) bf16 (roped)
    const unsigned short* __restrict__ Vc,  // (B, 8,T,64) bf16
    unsigned short* __restrict__ O) {       // (B,T,2048) bf16
  constexpr int T = 2048;
  __shared__ __attribute__((aligned(16))) unsigned short Ks[32 * 72];  // [k][d]
  __shared__ __attribute__((aligned(16))) unsigned short Vt[64 * 40];  // [d][k]
  __shared__ __attribute__((aligned(16))) unsigned short Ps[8 * 16 * 40];

  const int tid = threadIdx.x;
  const int lane = tid & 31;
  const int wave = tid >> 5;
  const int hl = lane >> 4;
  const int ll = lane & 15;
  const int qb0 = blockIdx.x * 128;
  const int h = blockIdx.y;
  const int b = blockIdx.z;
  const int hk = h >> 2;            // GQA: 4 q-heads per kv-head
  const int q0 = qb0 + wave * 16;

  // Q fragments (A-operand layout), held in registers for whole k-loop
  const unsigned short* qrow = Q + (((size_t)b * 32 + h) * T + q0 + ll) * 64;
  AF aq[2];
#pragma unroll
  for (int c = 0; c < 2; ++c) {
    aq[c].u[0] = *(const u32x4*)(qrow + c * 32 + 8 * hl);
    aq[c].u[1] = *(const u32x4*)(qrow + c * 32 + 16 + 8 * hl);
  }

  float mrow[8], lrow[8];
  v8f o[4];
#pragma unroll
  for (int r = 0; r < 8; ++r) { mrow[r] = -1e30f; lrow[r] = 0.0f; }
#pragma unroll
  for (int dn = 0; dn < 4; ++dn)
#pragma unroll
    for (int r = 0; r < 8; ++r) o[dn][r] = 0.0f;

  const unsigned short* kg = Kc + ((size_t)b * 8 + hk) * T * 64;
  const unsigned short* vg = Vc + ((size_t)b * 8 + hk) * T * 64;
  unsigned short* ps = Ps + wave * 16 * 40;

  const int krow_ld = tid >> 3;   // 0..31
  const int kseg = tid & 7;       // 8-half segment along d
  const int nsteps = (qb0 >> 5) + 4;

  for (int s = 0; s < nsteps; ++s) {
    const int kbase = s * 32;
    // K: async stage (row-major copy); V: transpose through registers
    async_g2l_b128(kg + (size_t)(kbase + krow_ld) * 64 + kseg * 8,
                   &Ks[krow_ld * 72 + kseg * 8]);
    u32x4 vv = *(const u32x4*)(vg + (size_t)(kbase + krow_ld) * 64 + kseg * 8);
    const unsigned short* vp = (const unsigned short*)&vv;
#pragma unroll
    for (int j = 0; j < 8; ++j) Vt[(kseg * 8 + j) * 40 + krow_ld] = vp[j];
    if (s + 1 < nsteps) {
      __builtin_prefetch(kg + (size_t)(kbase + 32 + krow_ld) * 64 + kseg * 8, 0, 1);
      __builtin_prefetch(vg + (size_t)(kbase + 32 + krow_ld) * 64 + kseg * 8, 0, 1);
    }
    wait_async0();
    __syncthreads();

    // S = Q K^T  (two 16-col tiles, contraction over d in two 32-chunks)
    v8f st[2];
#pragma unroll
    for (int j = 0; j < 2; ++j) {
#pragma unroll
      for (int r = 0; r < 8; ++r) st[j][r] = 0.0f;
#pragma unroll
      for (int c = 0; c < 2; ++c) {
        AF bkf;
        int krow = j * 16 + ll;
        bkf.u[0] = *(const u32x4*)&Ks[krow * 72 + c * 32 + 8 * hl];
        bkf.u[1] = *(const u32x4*)&Ks[krow * 72 + c * 32 + 16 + 8 * hl];
        st[j] = __builtin_amdgcn_wmma_f32_16x16x32_bf16(
            false, aq[c].v, false, bkf.v, (short)0, st[j], false, false);
      }
    }

    // online softmax (rows live per-VGPR across a 16-lane half)
    float p0[8], p1[8], scl[8];
#pragma unroll
    for (int r = 0; r < 8; ++r) {
      int qi = q0 + r + 8 * hl;
      float v0 = st[0][r] * 0.125f;
      float v1 = st[1][r] * 0.125f;
      if (kbase + ll > qi)      v0 = -1e30f;
      if (kbase + 16 + ll > qi) v1 = -1e30f;
      float rm = fmaxf(v0, v1);
      rm = fmaxf(rm, __shfl_xor(rm, 1));
      rm = fmaxf(rm, __shfl_xor(rm, 2));
      rm = fmaxf(rm, __shfl_xor(rm, 4));
      rm = fmaxf(rm, __shfl_xor(rm, 8));
      float mnew = fmaxf(mrow[r], rm);
      float sc = __expf(mrow[r] - mnew);
      float e0 = __expf(v0 - mnew);
      float e1 = __expf(v1 - mnew);
      float rs = e0 + e1;
      rs += __shfl_xor(rs, 1);
      rs += __shfl_xor(rs, 2);
      rs += __shfl_xor(rs, 4);
      rs += __shfl_xor(rs, 8);
      mrow[r] = mnew;
      lrow[r] = lrow[r] * sc + rs;
      scl[r] = sc;
      p0[r] = e0;
      p1[r] = e1;
    }

#pragma unroll
    for (int dn = 0; dn < 4; ++dn)
#pragma unroll
      for (int r = 0; r < 8; ++r) o[dn][r] *= scl[r];

    // P (C-layout) -> LDS -> A-layout fragment (same-wave DS ordering)
#pragma unroll
    for (int r = 0; r < 8; ++r) {
      int prow = r + 8 * hl;
      ps[prow * 40 + ll]      = f32_to_bf16_rne(p0[r]);
      ps[prow * 40 + 16 + ll] = f32_to_bf16_rne(p1[r]);
    }
    AF apf;
    apf.u[0] = *(const u32x4*)&ps[ll * 40 + 8 * hl];
    apf.u[1] = *(const u32x4*)&ps[ll * 40 + 16 + 8 * hl];

    // O += P V  (four d-column tiles, B-operand from transposed V)
#pragma unroll
    for (int dn = 0; dn < 4; ++dn) {
      AF bvf;
      int dcol = dn * 16 + ll;
      bvf.u[0] = *(const u32x4*)&Vt[dcol * 40 + 8 * hl];
      bvf.u[1] = *(const u32x4*)&Vt[dcol * 40 + 16 + 8 * hl];
      o[dn] = __builtin_amdgcn_wmma_f32_16x16x32_bf16(
          false, apf.v, false, bvf.v, (short)0, o[dn], false, false);
    }
    __syncthreads();
  }

#pragma unroll
  for (int r = 0; r < 8; ++r) lrow[r] = 1.0f / lrow[r];
#pragma unroll
  for (int dn = 0; dn < 4; ++dn)
#pragma unroll
    for (int r = 0; r < 8; ++r) {
      int qi = q0 + r + 8 * hl;
      O[((size_t)b * T + qi) * 2048 + h * 64 + dn * 16 + ll] =
          f32_to_bf16_rne(o[dn][r] * lrow[r]);
    }
}

// ---------------------------------------------------------------------------
extern "C" void kernel_launch(void* const* d_in, const int* in_sizes, int n_in,
                              void* d_out, int out_size, void* d_ws, size_t ws_size,
                              hipStream_t stream) {
  (void)in_sizes; (void)n_in; (void)out_size; (void)ws_size;
  const float* x  = (const float*)d_in[0];
  const float* Wq = (const float*)d_in[1];
  const float* Wk = (const float*)d_in[2];
  const float* Wv = (const float*)d_in[3];
  const float* Wo = (const float*)d_in[4];
  float* out = (float*)d_out;

  const int Bn = 2, T = 2048, H = 32, KVH = 8, D = 64;
  const int dim = H * D;        // 2048
  const int kvdim = KVH * D;    // 512
  const long long M = (long long)Bn * T;  // 4096

  char* ws = (char*)d_ws;
  size_t off = 0;
  auto alloc = [&](size_t bytes) -> char* {
    off = (off + 255) & ~(size_t)255;
    char* p = ws + off;
    off += bytes;
    return p;
  };

  unsigned short* xb   = (unsigned short*)alloc((size_t)M * dim * 2);
  unsigned short* Wqb  = (unsigned short*)alloc((size_t)dim * dim * 2);
  unsigned short* Wkb  = (unsigned short*)alloc((size_t)dim * kvdim * 2);
  unsigned short* Wvb  = (unsigned short*)alloc((size_t)dim * kvdim * 2);
  unsigned short* Wob  = (unsigned short*)alloc((size_t)dim * dim * 2);
  float* qf            = (float*)alloc((size_t)M * dim * 4);
  float* kf            = (float*)alloc((size_t)M * kvdim * 4);
  float* vf            = (float*)alloc((size_t)M * kvdim * 4);
  unsigned short* Qb   = (unsigned short*)alloc((size_t)Bn * H * T * D * 2);
  unsigned short* Kb2  = (unsigned short*)alloc((size_t)Bn * KVH * T * D * 2);
  unsigned short* Vb2  = (unsigned short*)alloc((size_t)Bn * KVH * T * D * 2);
  unsigned short* attb = (unsigned short*)alloc((size_t)M * dim * 2);

  cvt_f32_bf16_kernel<<<2048, 256, 0, stream>>>(x,  xb,  M * (long long)dim);
  cvt_f32_bf16_kernel<<<2048, 256, 0, stream>>>(Wq, Wqb, (long long)dim * dim);
  cvt_f32_bf16_kernel<<<512,  256, 0, stream>>>(Wk, Wkb, (long long)dim * kvdim);
  cvt_f32_bf16_kernel<<<512,  256, 0, stream>>>(Wv, Wvb, (long long)dim * kvdim);
  cvt_f32_bf16_kernel<<<2048, 256, 0, stream>>>(Wo, Wob, (long long)dim * dim);

  gemm_bf16_kernel<<<dim3(dim / 64, (unsigned)(M / 128)), 256, 0, stream>>>(
      xb, Wqb, qf, (int)M, dim, dim);
  gemm_bf16_kernel<<<dim3(kvdim / 64, (unsigned)(M / 128)), 256, 0, stream>>>(
      xb, Wkb, kf, (int)M, kvdim, dim);
  gemm_bf16_kernel<<<dim3(kvdim / 64, (unsigned)(M / 128)), 256, 0, stream>>>(
      xb, Wvb, vf, (int)M, kvdim, dim);

  long long qpairs = (long long)Bn * T * H * (D / 2);
  long long kpairs = (long long)Bn * T * KVH * (D / 2);
  rope_permute_kernel<<<4096, 256, 0, stream>>>(qf, Qb, H, T, 1, qpairs);
  rope_permute_kernel<<<1024, 256, 0, stream>>>(kf, Kb2, KVH, T, 1, kpairs);
  rope_permute_kernel<<<1024, 256, 0, stream>>>(vf, Vb2, KVH, T, 0, kpairs);

  attn_kernel<<<dim3(T / 128, H, Bn), 256, 0, stream>>>(Qb, Kb2, Vb2, attb);

  gemm_bf16_kernel<<<dim3(dim / 64, (unsigned)(M / 128)), 256, 0, stream>>>(
      attb, Wob, out, (int)M, dim, dim);
}
